// SingleStreamAdvKPCN_15109694947675
// MI455X (gfx1250) — matrix operand order repaired and checked by reading, not compiled, not run on hardware
//
#include <hip/hip_runtime.h>

// ---------------------------------------------------------------------------
// CDNA5 (gfx1250) KPCN denoiser.
// Convs = implicit GEMM on V_WMMA_F32_16X16X32_BF16, A-tile staged to LDS via
// global_load_async_to_lds_b128 (ASYNCcnt) with double buffering.
// ---------------------------------------------------------------------------

typedef __attribute__((ext_vector_type(16))) __bf16 v16bf;
typedef __attribute__((ext_vector_type(8)))  __bf16 v8bf;
typedef __attribute__((ext_vector_type(8)))  float  v8f;

__device__ __forceinline__ __bf16 f2bf(float f) {
    unsigned u = __builtin_bit_cast(unsigned, f);
    unsigned r = (u + 0x7FFFu + ((u >> 16) & 1u)) >> 16;   // round-to-nearest-even
    unsigned short s = (unsigned short)r;
    return __builtin_bit_cast(__bf16, s);
}

// ---- weight conversion: f32 [OC][K] -> bf16 [OC][Kpad] (zero tail) ---------
__global__ void cvt_w(const float* __restrict__ w, __bf16* __restrict__ o,
                      int OC, int K, int Kpad) {
    int i = blockIdx.x * blockDim.x + threadIdx.x;
    if (i >= OC * Kpad) return;
    int oc = i / Kpad, k = i - oc * Kpad;
    o[i] = (k < K) ? f2bf(w[(size_t)oc * K + k]) : f2bf(0.0f);
}

// ---- async-stage one 16x32 bf16 A-tile (1KB) into LDS ----------------------
// Waves 0 and 1 (t<64) each issue one global_load_async_to_lds_b128 (16B/lane).
__device__ __forceinline__ void stage_A(const __bf16* wbf, int Kpad, int ocBase,
                                        int OC, int kb, __bf16* dst /*LDS*/) {
    int t = threadIdx.x;
    if (t < 64) {                                   // waves 0,1 -> wave-uniform
        int row = t >> 2, ch = t & 3;               // 16 rows x 4 chunks of 8 bf16
        int mrow = ocBase + row; if (mrow >= OC) mrow = OC - 1;
        const __bf16* gp = wbf + (size_t)mrow * Kpad + kb + ch * 8;
        unsigned lo = (unsigned)(uintptr_t)(dst + row * 32 + ch * 8); // LDS byte off
        asm volatile("global_load_async_to_lds_b128 %0, %1, off"
                     :: "v"(lo), "v"(gp) : "memory");
    }
}

// ---- implicit-GEMM conv (stride 1). Block: 16 OC x 512 spatial.
//      Wave: 16 OC x 64 spatial = 4 WMMA accumulators sharing one A fragment.
__global__ __launch_bounds__(256)
void conv_wmma(const float* __restrict__ x, const __bf16* __restrict__ wbf,
               const float* __restrict__ bias, float* __restrict__ y,
               int Cin, int Hin, int Win, int OC, int Hout, int Wout,
               int KH, int KW, int pad, int K, int Kpad, int relu) {
    __shared__ __align__(16) __bf16 As[2][16 * 32];  // double-buffered A tile

    const int lane = threadIdx.x & 31;
    const int hl   = lane >> 4;                      // wave32 lane-half
    const int wid  = threadIdx.x >> 5;               // wave in block (0..7)
    const int P = Hout * Wout;
    const int tilesM = (OC + 15) >> 4;
    const int tm     = blockIdx.x % tilesM;
    const int ocBase = tm << 4;
    const int pBlock = (blockIdx.x / tilesM) * 512;
    const int khkw   = KH * KW;

    // 4 column groups of 16 spatial positions for this wave
    int  colv[4], ohv[4], owv[4];
    bool okv[4];
    #pragma unroll
    for (int q = 0; q < 4; ++q) {
        int col  = pBlock + wid * 64 + q * 16 + (lane & 15);
        bool ok  = col < P;
        colv[q] = col; okv[q] = ok;
        ohv[q]  = ok ? col / Wout : 0;
        owv[q]  = ok ? col - ohv[q] * Wout : 0;
    }

    v8f acc[4] = {};
    stage_A(wbf, Kpad, ocBase, OC, 0, &As[0][0]);
    int ib = 0;
    for (int kb = 0; kb < Kpad; kb += 32, ib ^= 1) {
        if (kb + 32 < Kpad) {                        // prefetch next A tile
            stage_A(wbf, Kpad, ocBase, OC, kb + 32, &As[ib ^ 1][0]);
            asm volatile("s_wait_asynccnt 0x1" ::: "memory");
        } else {
            asm volatile("s_wait_asynccnt 0x0" ::: "memory");
        }
        __syncthreads();                             // publish As[ib]

        // A fragment from LDS (ISA 7.12.2 bf16 16x32 layout)
        const __bf16* ap = &As[ib][(lane & 15) * 32];
        v8bf alo = *reinterpret_cast<const v8bf*>(ap + hl * 8);
        v8bf ahi = *reinterpret_cast<const v8bf*>(ap + 16 + hl * 8);
        v16bf a;
        #pragma unroll
        for (int i = 0; i < 8; ++i) { a[i] = alo[i]; a[i + 8] = ahi[i]; }

        // 4 B fragments; (ic,kh,kw) decomposition shared across the 4 groups
        v16bf b[4];
        #pragma unroll
        for (int e = 0; e < 16; ++e) {
            int k = kb + hl * 8 + (e & 7) + ((e >> 3) << 4);
            bool kOK = (k < K);
            int ic = 0, kh = 0, kw = 0;
            if (kOK) {
                ic = k / khkw;
                int r = k - ic * khkw;
                kh = r / KW;
                kw = r - kh * KW;
            }
            #pragma unroll
            for (int q = 0; q < 4; ++q) {
                float v = 0.0f;
                if (kOK && okv[q]) {
                    int ih = ohv[q] - pad + kh;
                    int iw = owv[q] - pad + kw;
                    if ((unsigned)ih < (unsigned)Hin && (unsigned)iw < (unsigned)Win)
                        v = x[((size_t)ic * Hin + ih) * Win + iw];
                }
                b[q][e] = f2bf(v);
            }
        }
        acc[0] = __builtin_amdgcn_wmma_f32_16x16x32_bf16(false, a, false, b[0], (short)0, acc[0], false, false);
        acc[1] = __builtin_amdgcn_wmma_f32_16x16x32_bf16(false, a, false, b[1], (short)0, acc[1], false, false);
        acc[2] = __builtin_amdgcn_wmma_f32_16x16x32_bf16(false, a, false, b[2], (short)0, acc[2], false, false);
        acc[3] = __builtin_amdgcn_wmma_f32_16x16x32_bf16(false, a, false, b[3], (short)0, acc[3], false, false);
        __syncthreads();                             // reads done before re-stage
    }

    // C layout: M = r + 8*half, N = lane&15
    #pragma unroll
    for (int q = 0; q < 4; ++q) {
        if (!okv[q]) continue;
        #pragma unroll
        for (int r = 0; r < 8; ++r) {
            int oc = ocBase + r + hl * 8;
            if (oc < OC) {
                float v = acc[q][r] + bias[oc];
                if (relu) v = v > 0.0f ? v : 0.0f;
                y[(size_t)oc * P + colv[q]] = v;
            }
        }
    }
}

// ---- channel softmax over C at each of P pixels ----------------------------
__global__ void softmax_ch(const float* __restrict__ in, float* __restrict__ out,
                           int C, int P) {
    int p = blockIdx.x * blockDim.x + threadIdx.x;
    if (p >= P) return;
    float m = -3.4e38f;
    for (int c = 0; c < C; ++c) { float v = in[(size_t)c * P + p]; m = v > m ? v : m; }
    float s = 0.0f;
    for (int c = 0; c < C; ++c) s += __expf(in[(size_t)c * P + p] - m);
    float inv = 1.0f / s;
    for (int c = 0; c < C; ++c)
        out[(size_t)c * P + p] = __expf(in[(size_t)c * P + p] - m) * inv;
}

// ---- 21x21 kernel apply with implicit center crop of the buffer ------------
__global__ void kapply(const float* __restrict__ buf, int bufH, int bufW, int off,
                       const float* __restrict__ wgt, float* __restrict__ out,
                       int C, int H, int W) {
    int idx = blockIdx.x * blockDim.x + threadIdx.x;
    int P = H * W;
    if (idx >= C * P) return;
    int c = idx / P, p = idx - c * P, h = p / W, w = p - h * W;
    float s = 0.0f;
    for (int kh = 0; kh < 21; ++kh) {
        int yy = h - 10 + kh;
        if ((unsigned)yy >= (unsigned)H) continue;
        for (int kw = 0; kw < 21; ++kw) {
            int xx = w - 10 + kw;
            if ((unsigned)xx >= (unsigned)W) continue;
            s += buf[((size_t)c * bufH + yy + off) * bufW + xx + off]
               * wgt[(size_t)(kh * 21 + kw) * P + p];
        }
    }
    out[idx] = s;
}

// ---- elementwise / data-movement helpers -----------------------------------
__global__ void ew_expm1(const float* x, float* y, int n) {
    int i = blockIdx.x * blockDim.x + threadIdx.x;
    if (i < n) y[i] = __expf(x[i]) - 1.0f;
}
__global__ void ew_sigmoid(const float* x, float* y, int n) {
    int i = blockIdx.x * blockDim.x + threadIdx.x;
    if (i < n) y[i] = 1.0f / (1.0f + __expf(-x[i]));
}
__global__ void fma_bcast(const float* a, const float* g, const float* f,
                          float* o, int C, int P) {
    int i = blockIdx.x * blockDim.x + threadIdx.x;
    if (i >= C * P) return;
    int p = i % P;
    o[i] = a[i] + g[i] * f[p];
}
__global__ void copy_crop(const float* __restrict__ src, int sH, int sW, int off,
                          int chStart, float* __restrict__ dst, int H, int W,
                          int dstChStart, int nCh) {
    int i = blockIdx.x * blockDim.x + threadIdx.x;
    if (i >= nCh * H * W) return;
    int c = i / (H * W), p = i - c * (H * W), h = p / W, w = p - h * W;
    dst[(size_t)(dstChStart + c) * H * W + p] =
        src[((size_t)(chStart + c) * sH + h + off) * sW + w + off];
}
__global__ void avgpool2(const float* __restrict__ x, float* __restrict__ y,
                         int C, int H, int W) {
    int Ho = H >> 1, Wo = W >> 1;
    int i = blockIdx.x * blockDim.x + threadIdx.x;
    if (i >= C * Ho * Wo) return;
    int c = i / (Ho * Wo), p = i - c * (Ho * Wo), h = p / Wo, w = p - h * Wo;
    const float* s = x + ((size_t)c * H + 2 * h) * W + 2 * w;
    y[i] = 0.25f * (s[0] + s[1] + s[W] + s[W + 1]);
}
__global__ void upcat(const float* __restrict__ bt, const float* __restrict__ e1,
                      float* __restrict__ dst, int Cb, int Ce, int H, int W) {
    int C = Cb + Ce;
    int i = blockIdx.x * blockDim.x + threadIdx.x;
    if (i >= C * H * W) return;
    int c = i / (H * W), p = i - c * (H * W), h = p / W, w = p - h * W;
    if (c < Cb)
        dst[i] = bt[((size_t)c * (H >> 1) + (h >> 1)) * (W >> 1) + (w >> 1)];
    else
        dst[i] = e1[(size_t)(c - Cb) * H * W + p];
}

// ---------------------------------------------------------------------------
// Host orchestration
// ---------------------------------------------------------------------------
static inline int g1(int n) { return (n + 255) / 256; }

static void launch_conv(hipStream_t s, const float* x, const __bf16* w, const float* b,
                        float* y, int Cin, int Hin, int Win, int OC, int Hout, int Wout,
                        int KH, int KW, int pad, int K, int Kpad, int relu) {
    long tilesM  = (OC + 15) / 16;
    long blocksN = ((long)Hout * Wout + 511) / 512;
    int blocks = (int)(tilesM * blocksN);
    conv_wmma<<<blocks, 256, 0, s>>>(x, w, b, y, Cin, Hin, Win, OC, Hout, Wout,
                                     KH, KW, pad, K, Kpad, relu);
}

extern "C" void kernel_launch(void* const* d_in, const int* in_sizes, int n_in,
                              void* d_out, int out_size, void* d_ws, size_t ws_size,
                              hipStream_t stream) {
    const float* kpcn_in  = (const float*)d_in[0];
    const float* kpcn_buf = (const float*)d_in[1];
    const float* target   = (const float*)d_in[2];
    const float* g_w0 = (const float*)d_in[3];  const float* g_b0 = (const float*)d_in[4];
    const float* g_wm = (const float*)d_in[5];  const float* g_bm = (const float*)d_in[6];
    const float* g_wo = (const float*)d_in[7];  const float* g_bo = (const float*)d_in[8];
    const float* p_w0 = (const float*)d_in[9];  const float* p_b0 = (const float*)d_in[10];
    const float* p_wm = (const float*)d_in[11]; const float* p_bm = (const float*)d_in[12];
    const float* p_wo = (const float*)d_in[13]; const float* p_bo = (const float*)d_in[14];
    const float* u_w1 = (const float*)d_in[15]; const float* u_b1 = (const float*)d_in[16];
    const float* u_w2 = (const float*)d_in[17]; const float* u_b2 = (const float*)d_in[18];
    const float* u_w3 = (const float*)d_in[19]; const float* u_b3 = (const float*)d_in[20];
    const float* u_w4 = (const float*)d_in[21]; const float* u_b4 = (const float*)d_in[22];

    const int P156 = 156 * 156, P192 = 192 * 192;
    float* out = (float*)d_out;
    float* O_final = out;
    float* O_gbuf  = O_final + 3 * P156;
    float* O_fake  = O_gbuf  + 3 * P156;
    float* O_real  = O_fake  + P156;
    float* O_kern  = O_real  + P192;
    float* O_kgbuf = O_kern  + 441 * P156;

    // ---- workspace bump allocator ----
    char*  base = (char*)d_ws;
    size_t off  = 0;
    auto alloc = [&](size_t bytes) -> char* {
        char* p = base + off;
        off += (bytes + 255) & ~(size_t)255;
        return p;
    };
    __bf16* Wg0 = (__bf16*)alloc((size_t)100 * 704  * 2);
    __bf16* Wgm[7]; for (int i = 0; i < 7; ++i) Wgm[i] = (__bf16*)alloc((size_t)100 * 2528 * 2);
    __bf16* Wgo = (__bf16*)alloc((size_t)441 * 2528 * 2);
    __bf16* Wp0 = (__bf16*)alloc((size_t)100 * 480  * 2);
    __bf16* Wpm[7]; for (int i = 0; i < 7; ++i) Wpm[i] = (__bf16*)alloc((size_t)100 * 928 * 2);
    __bf16* Wpo = (__bf16*)alloc((size_t)441 * 928  * 2);
    __bf16* Wu1 = (__bf16*)alloc((size_t)64  * 256  * 2);
    __bf16* Wu2 = (__bf16*)alloc((size_t)128 * 576  * 2);
    __bf16* Wu3 = (__bf16*)alloc((size_t)128 * 1152 * 2);
    __bf16* Wu4 = (__bf16*)alloc((size_t)51  * 1728 * 2);
    float* bufA = (float*)alloc((size_t)441 * P156 * 4);
    float* bufB = (float*)alloc((size_t)441 * P156 * 4);
    float* e1b  = (float*)alloc((size_t)64  * P192 * 4);
    float* pb   = (float*)alloc((size_t)64  * 96 * 96 * 4);
    float* e2b  = (float*)alloc((size_t)128 * 96 * 96 * 4);
    float* btb  = (float*)alloc((size_t)128 * 96 * 96 * 4);
    float* catb = (float*)alloc((size_t)192 * P192 * 4);
    float* xb   = (float*)alloc((size_t)51 * P156 * 4);
    float* yb   = (float*)alloc((size_t)51 * P192 * 4);
    float* gr   = (float*)alloc((size_t)3 * P156 * 4);
    float* rad  = (float*)alloc((size_t)3 * P156 * 4);

    // ---- weight conversion to padded bf16 ----
    cvt_w<<<g1(100 * 704), 256, 0, stream>>>(g_w0, Wg0, 100, 675, 704);
    for (int i = 0; i < 7; ++i)
        cvt_w<<<g1(100 * 2528), 256, 0, stream>>>(g_wm + (size_t)i * 250000, Wgm[i], 100, 2500, 2528);
    cvt_w<<<g1(441 * 2528), 256, 0, stream>>>(g_wo, Wgo, 441, 2500, 2528);
    cvt_w<<<g1(100 * 480), 256, 0, stream>>>(p_w0, Wp0, 100, 450, 480);
    for (int i = 0; i < 7; ++i)
        cvt_w<<<g1(100 * 928), 256, 0, stream>>>(p_wm + (size_t)i * 90000, Wpm[i], 100, 900, 928);
    cvt_w<<<g1(441 * 928), 256, 0, stream>>>(p_wo, Wpo, 441, 900, 928);
    cvt_w<<<g1(64  * 256), 256, 0, stream>>>(u_w1, Wu1, 64, 243, 256);
    cvt_w<<<g1(128 * 576), 256, 0, stream>>>(u_w2, Wu2, 128, 576, 576);
    cvt_w<<<g1(128 * 1152), 256, 0, stream>>>(u_w3, Wu3, 128, 1152, 1152);
    cvt_w<<<g1(51  * 1728), 256, 0, stream>>>(u_w4, Wu4, 51, 1728, 1728);

    // ---- g-chain: 9x VALID 5x5, 192 -> 156 ----
    launch_conv(stream, kpcn_in, Wg0, g_b0, bufA, 27, 192, 192, 100, 188, 188, 5, 5, 0, 675, 704, 1);
    float* cur = bufA; float* nxt = bufB; int h = 188;
    for (int i = 0; i < 7; ++i) {
        launch_conv(stream, cur, Wgm[i], g_bm + i * 100, nxt, 100, h, h, 100, h - 4, h - 4, 5, 5, 0, 2500, 2528, 1);
        float* t = cur; cur = nxt; nxt = t; h -= 4;
    }                                                  // cur holds (100,160,160)
    launch_conv(stream, cur, Wgo, g_bo, O_kgbuf, 100, 160, 160, 441, 156, 156, 5, 5, 0, 2500, 2528, 0);

    // ---- softmax + kernel apply -> gbuf_r, gbuf_final ----
    softmax_ch<<<g1(P156), 256, 0, stream>>>(O_kgbuf, bufA, 441, P156);
    kapply<<<g1(3 * P156), 256, 0, stream>>>(kpcn_buf, 192, 192, 18, bufA, gr, 3, 156, 156);
    ew_expm1<<<g1(3 * P156), 256, 0, stream>>>(gr, O_gbuf, 3 * P156);

    // ---- fake_in = [gbuf_r, crop(kpcn_in[10:34])] ----
    copy_crop<<<g1(3 * P156), 256, 0, stream>>>(gr, 156, 156, 0, 0, bufB, 156, 156, 0, 3);
    copy_crop<<<g1(24 * P156), 256, 0, stream>>>(kpcn_in, 192, 192, 18, 10, bufB, 156, 156, 3, 24);

    // ---- U-Net (fake, 156) ----
    launch_conv(stream, bufB, Wu1, u_b1, e1b, 27, 156, 156, 64, 156, 156, 3, 3, 1, 243, 256, 1);
    avgpool2<<<g1(64 * 78 * 78), 256, 0, stream>>>(e1b, pb, 64, 156, 156);
    launch_conv(stream, pb, Wu2, u_b2, e2b, 64, 78, 78, 128, 78, 78, 3, 3, 1, 576, 576, 1);
    launch_conv(stream, e2b, Wu3, u_b3, btb, 128, 78, 78, 128, 78, 78, 3, 3, 1, 1152, 1152, 1);
    upcat<<<g1(192 * P156), 256, 0, stream>>>(btb, e1b, catb, 128, 64, 156, 156);
    launch_conv(stream, catb, Wu4, u_b4, xb, 192, 156, 156, 51, 156, 156, 3, 3, 1, 1728, 1728, 0);
    ew_sigmoid<<<g1(P156), 256, 0, stream>>>(xb, O_fake, P156);

    // ---- U-Net (real, 192) ----
    copy_crop<<<g1(3 * P192), 256, 0, stream>>>(target, 192, 192, 0, 0, bufB, 192, 192, 0, 3);
    copy_crop<<<g1(24 * P192), 256, 0, stream>>>(kpcn_in, 192, 192, 0, 10, bufB, 192, 192, 3, 24);
    launch_conv(stream, bufB, Wu1, u_b1, e1b, 27, 192, 192, 64, 192, 192, 3, 3, 1, 243, 256, 1);
    avgpool2<<<g1(64 * 96 * 96), 256, 0, stream>>>(e1b, pb, 64, 192, 192);
    launch_conv(stream, pb, Wu2, u_b2, e2b, 64, 96, 96, 128, 96, 96, 3, 3, 1, 576, 576, 1);
    launch_conv(stream, e2b, Wu3, u_b3, btb, 128, 96, 96, 128, 96, 96, 3, 3, 1, 1152, 1152, 1);
    upcat<<<g1(192 * P192), 256, 0, stream>>>(btb, e1b, catb, 128, 64, 192, 192);
    launch_conv(stream, catb, Wu4, u_b4, yb, 192, 192, 192, 51, 192, 192, 3, 3, 1, 1728, 1728, 0);
    ew_sigmoid<<<g1(P192), 256, 0, stream>>>(yb, O_real, P192);

    // ---- p-chain on feat = x[:,1:] (50,156,156) ----
    const float* feat = xb + P156;
    launch_conv(stream, feat, Wp0, p_b0, bufA, 50, 156, 156, 100, 156, 156, 3, 3, 1, 450, 480, 1);
    cur = bufA; nxt = bufB;
    for (int i = 0; i < 7; ++i) {
        launch_conv(stream, cur, Wpm[i], p_bm + i * 100, nxt, 100, 156, 156, 100, 156, 156, 3, 3, 1, 900, 928, 1);
        float* t = cur; cur = nxt; nxt = t;
    }                                                  // cur = bufB after 7 swaps
    launch_conv(stream, cur, Wpo, p_bo, nxt, 100, 156, 156, 441, 156, 156, 3, 3, 1, 900, 928, 0);
    float* pout = nxt;                                 // (441,156,156)

    // ---- kernel = pout + k_gbuf * fake ; double softmax ; apply ; final ----
    fma_bcast<<<g1(441 * P156), 256, 0, stream>>>(pout, O_kgbuf, O_fake, O_kern, 441, P156);
    float* sA = (pout == bufA) ? bufB : bufA;
    softmax_ch<<<g1(P156), 256, 0, stream>>>(O_kern, sA, 441, P156);
    softmax_ch<<<g1(P156), 256, 0, stream>>>(sA, pout, 441, P156);
    kapply<<<g1(3 * P156), 256, 0, stream>>>(kpcn_buf, 192, 192, 18, pout, rad, 3, 156, 156);
    ew_expm1<<<g1(3 * P156), 256, 0, stream>>>(rad, O_final, 3 * P156);
    (void)in_sizes; (void)n_in; (void)out_size; (void)ws_size;
}
